// FeedForward_MoE_76063870812416
// MI455X (gfx1250) — matrix operand order repaired and checked by reading, not compiled, not run on hardware
//
#include <hip/hip_runtime.h>
#include <hip/hip_bf16.h>
#include <stdint.h>

// Problem constants (from reference): B=4, T=2048, C=1024, E=8, K=2, H=4096, CAP=2048
#define B_     4
#define T_     2048
#define C_     1024
#define E_     8
#define KTOP_  2
#define H_     4096
#define CAP_   2048
#define NTOK_  (B_ * T_)   // 8192 tokens

typedef unsigned short u16;
typedef __attribute__((ext_vector_type(16))) __bf16 bf16x16;
typedef __attribute__((ext_vector_type(8)))  float  floatx8;

struct Pack32 { uint4 lo, hi; };          // 32 bytes == 16 bf16 (one WMMA fragment)
struct Half4  { u16 a, b, c, d; };        // 8 bytes

__device__ __forceinline__ float b2f(u16 u) {
  unsigned int x = ((unsigned int)u) << 16;
  return __builtin_bit_cast(float, x);
}
__device__ __forceinline__ u16 f2b(float f) {
  unsigned int x = __builtin_bit_cast(unsigned int, f);
  unsigned int r = x + 0x7FFFu + ((x >> 16) & 1u);   // round-to-nearest-even
  return (u16)(r >> 16);
}

// CDNA5 async copy: global -> LDS, 16B per lane, tracked by ASYNCcnt.
// (GLOBAL_LOAD_ASYNC_TO_LDS_B128, ISA 15.18.3 op 98; no VGPR round-trip.)
__device__ __forceinline__ void async_copy_b128(unsigned lds_off, const void* gptr) {
  asm volatile("global_load_async_to_lds_b128 %0, %1, off"
               :: "v"(lds_off), "v"(gptr)
               : "memory");
}
__device__ __forceinline__ void wait_async_le4() {
  asm volatile("s_wait_asynccnt 0x4" ::: "memory");
}
__device__ __forceinline__ void wait_async_le0() {
  asm volatile("s_wait_asynccnt 0x0" ::: "memory");
}

// ---------------------------------------------------------------------------
// f32 -> bf16 conversion (vectorized: 4 floats -> 8 bytes per thread)
// ---------------------------------------------------------------------------
__global__ void cvt_f32_bf16(const float* __restrict__ src, u16* __restrict__ dst, int n4) {
  int i = blockIdx.x * blockDim.x + threadIdx.x;
  if (i >= n4) return;
  float4 f = ((const float4*)src)[i];
  Half4 h { f2b(f.x), f2b(f.y), f2b(f.z), f2b(f.w) };
  ((uint2*)dst)[i] = __builtin_bit_cast(uint2, h);
}

// ---------------------------------------------------------------------------
// Weight prep: f32 [K][N] -> bf16 [N][K] (transpose fused into conversion).
// K-contiguous weights let BOTH GEMM operands stream via async-to-LDS b128.
// 32x32 tiles, 256 threads (32x8), padded LDS, batched via blockIdx.z.
// ---------------------------------------------------------------------------
__global__ void __launch_bounds__(256)
cvt_transpose_f32_bf16(const float* __restrict__ src, u16* __restrict__ dst,
                       int Kd, int Nd) {
  __shared__ float tile[32][33];
  const size_t zoff = (size_t)blockIdx.z * (size_t)Kd * (size_t)Nd;
  src += zoff;
  dst += zoff;
  const int k0 = blockIdx.y * 32, n0 = blockIdx.x * 32;
  const int tx = threadIdx.x & 31, ty = threadIdx.x >> 5;   // 32 x 8
#pragma unroll
  for (int i = 0; i < 32; i += 8)
    tile[ty + i][tx] = src[(size_t)(k0 + ty + i) * Nd + (n0 + tx)];
  __syncthreads();
#pragma unroll
  for (int i = 0; i < 32; i += 8)
    dst[(size_t)(n0 + ty + i) * Kd + (k0 + tx)] = f2b(tile[tx][ty + i]);
}

__global__ void zero_u32(unsigned int* __restrict__ p, int n) {
  int i = blockIdx.x * blockDim.x + threadIdx.x;
  if (i < n) p[i] = 0u;
}

// ---------------------------------------------------------------------------
// Tiled bf16 WMMA GEMM:  C[M,N] = act(A[M,K] * Bt[N,K]^T + bias[N])
//   - 256 threads = 8 wave32s; block tile 128x128; wave tile 64x32
//   - double-buffered LDS; tiles streamed with global_load_async_to_lds_b128
//     (4 async ops/thread/stage -> exact s_wait_asynccnt pipelining)
//   - fragments: two ds_read_b128 of contiguous K per lane, matching the
//     CDNA5 16-bit A/B lane layouts (ISA 7.12.2)
//   - batched over blockIdx.z with element strides (for the 8 experts)
// ---------------------------------------------------------------------------
template <bool RELU, bool OUTBF16>
__global__ void __launch_bounds__(256)
gemm_bf16_wmma(const u16* __restrict__ A, const u16* __restrict__ Bt,
               const float* __restrict__ bias, void* __restrict__ Cout,
               int Md, int Nd, int Kd,
               long long strideA, long long strideB,
               long long strideBias, long long strideC)
{
  __shared__ u16 As[2][128][40];   // 32 K-halves + 8 pad (16B-aligned rows)
  __shared__ u16 Bs[2][128][40];   // [n][k], same shape as A

  const int tid   = threadIdx.x;
  const int lane  = tid & 31;
  const int wave  = tid >> 5;
  const int waveM = wave >> 2;          // 0..1
  const int waveN = wave & 3;           // 0..3
  const int m0    = blockIdx.y * 128;
  const int n0    = blockIdx.x * 128;
  const int z     = blockIdx.z;

  A    += (size_t)z * (size_t)strideA;
  Bt   += (size_t)z * (size_t)strideB;
  bias += (size_t)z * (size_t)strideBias;

  floatx8 zero8 = {0.f, 0.f, 0.f, 0.f, 0.f, 0.f, 0.f, 0.f};
  floatx8 acc[4][2];
#pragma unroll
  for (int i = 0; i < 4; ++i)
#pragma unroll
    for (int j = 0; j < 2; ++j) acc[i][j] = zero8;

  // Per-thread staging coordinates: 512 chunks of 8 halves per 128x32 tile.
  // Each thread issues 2 async copies for A + 2 for B per stage (uniform).
  const int sr0 = tid >> 2;                 // rows for it=0 / it=1
  const int sr1 = (tid + 256) >> 2;
  const int sc  = (tid & 3) << 3;           // 0,8,16,24

  auto stage = [&](int step, int buf) {
    const int k0 = step << 5;
    async_copy_b128((unsigned)(uintptr_t)&As[buf][sr0][sc],
                    A + (size_t)(m0 + sr0) * Kd + (k0 + sc));
    async_copy_b128((unsigned)(uintptr_t)&As[buf][sr1][sc],
                    A + (size_t)(m0 + sr1) * Kd + (k0 + sc));
    async_copy_b128((unsigned)(uintptr_t)&Bs[buf][sr0][sc],
                    Bt + (size_t)(n0 + sr0) * Kd + (k0 + sc));
    async_copy_b128((unsigned)(uintptr_t)&Bs[buf][sr1][sc],
                    Bt + (size_t)(n0 + sr1) * Kd + (k0 + sc));
  };

  // CDNA5 16-bit fragment lane layout (ISA 7.12.2):
  //  A 16x32: lane<16 -> halves {K 0..7, K 16..23}; lane>=16 -> {K 8..15, K 24..31}
  //  B 32x16: lane<16 -> halves {K 0..7, K 16..23}; lane>=16 -> {K 8..15, K 24..31}
  const int kgA = (lane < 16) ? 0 : 8;
  const int ksB = (lane < 16) ? 0 : 8;

  const int nsteps = Kd >> 5;
  stage(0, 0);                               // prologue: tile 0 -> buf 0

  for (int s = 0; s < nsteps; ++s) {
    const int cur = s & 1;
    const bool hasNext = (s + 1) < nsteps;
    if (hasNext) {
      stage(s + 1, cur ^ 1);                 // stream next tile while computing
      wait_async_le4();                      // cur's 4 copies retired (in-order)
    } else {
      wait_async_le0();
    }
    __syncthreads();

    bf16x16 afrag[4];
#pragma unroll
    for (int mi = 0; mi < 4; ++mi) {
      const u16* rp = &As[cur][waveM * 64 + mi * 16 + (lane & 15)][0];
      Pack32 p;
      p.lo = *(const uint4*)(rp + kgA);
      p.hi = *(const uint4*)(rp + 16 + kgA);
      afrag[mi] = __builtin_bit_cast(bf16x16, p);
    }
    bf16x16 bfrag[2];
#pragma unroll
    for (int ni = 0; ni < 2; ++ni) {
      const u16* cp = &Bs[cur][waveN * 32 + ni * 16 + (lane & 15)][0];
      Pack32 p;
      p.lo = *(const uint4*)(cp + ksB);
      p.hi = *(const uint4*)(cp + ksB + 16);
      bfrag[ni] = __builtin_bit_cast(bf16x16, p);
    }
#pragma unroll
    for (int mi = 0; mi < 4; ++mi)
#pragma unroll
      for (int ni = 0; ni < 2; ++ni)
        acc[mi][ni] = __builtin_amdgcn_wmma_f32_16x16x32_bf16(
            false, afrag[mi], false, bfrag[ni], (short)0, acc[mi][ni], false, false);

    __syncthreads();                         // all reads of 'cur' done before restage
  }

  // ---- epilogue: bias + optional ReLU, store bf16 or f32 ------------------
  u16*   outb = (u16*)Cout   + (size_t)z * (size_t)strideC;
  float* outf = (float*)Cout + (size_t)z * (size_t)strideC;
  const int mofs = (lane < 16) ? 0 : 8;     // C/D layout: lanes 0-15 -> M=v, 16-31 -> M=v+8
#pragma unroll
  for (int mi = 0; mi < 4; ++mi) {
#pragma unroll
    for (int ni = 0; ni < 2; ++ni) {
      int nn = n0 + waveN * 32 + ni * 16 + (lane & 15);
      int mb = m0 + waveM * 64 + mi * 16 + mofs;
      float bv = bias[nn];
#pragma unroll
      for (int v = 0; v < 8; ++v) {
        float val = acc[mi][ni][v] + bv;
        if (RELU) val = fmaxf(val, 0.0f);
        size_t off = (size_t)(mb + v) * (size_t)Nd + (size_t)nn;
        if (OUTBF16) outb[off] = f2b(val);
        else         outf[off] = val;
      }
    }
  }
  (void)Md;
}

// ---------------------------------------------------------------------------
// Gating: logits = h2 @ rw3 + rb3 (N x 8), softmax, top-2 (ties -> lower idx)
// One block per token: 8 experts x 32 K-slices.
// ---------------------------------------------------------------------------
__global__ void __launch_bounds__(256)
gate_topk(const u16* __restrict__ h2, const float* __restrict__ rw3,
          const float* __restrict__ rb3, int* __restrict__ topk_id,
          float* __restrict__ topk_p)
{
  const int n   = blockIdx.x;
  const int tid = threadIdx.x;
  const int e   = tid & 7;
  const int sl  = tid >> 3;                  // 0..31
  __shared__ float red[256];
  __shared__ float logits[E_];

  float acc = 0.f;
  const u16* hp = h2 + (size_t)n * H_;
  for (int k = sl; k < H_; k += 32)
    acc += b2f(hp[k]) * rw3[k * E_ + e];
  red[tid] = acc;
  __syncthreads();
  for (int s = 128; s >= 8; s >>= 1) {
    if (tid < s) red[tid] += red[tid + s];
    __syncthreads();
  }
  if (tid < E_) logits[tid] = red[tid] + rb3[tid];
  __syncthreads();
  if (tid == 0) {
    float mx = logits[0];
    for (int i = 1; i < E_; ++i) mx = fmaxf(mx, logits[i]);
    float p[E_], ssum = 0.f;
    for (int i = 0; i < E_; ++i) { p[i] = __expf(logits[i] - mx); ssum += p[i]; }
    float inv = 1.f / ssum;
    int i0 = 0;
    for (int i = 1; i < E_; ++i) if (p[i] > p[i0]) i0 = i;
    int i1 = -1;
    for (int i = 0; i < E_; ++i) {
      if (i == i0) continue;
      if (i1 < 0 || p[i] > p[i1]) i1 = i;
    }
    topk_id[2 * n + 0] = i0;  topk_p[2 * n + 0] = p[i0] * inv;
    topk_id[2 * n + 1] = i1;  topk_p[2 * n + 1] = p[i1] * inv;
  }
}

// ---------------------------------------------------------------------------
// Deterministic flat-order per-expert positions (matches reference cumsum).
// Single workgroup; 16384 elements in 64 chunks of 256 with an LDS scan.
// ---------------------------------------------------------------------------
__global__ void __launch_bounds__(256)
compute_pos(const int* __restrict__ eid, int* __restrict__ pos, int total)
{
  __shared__ int se[256];
  __shared__ int base[E_];
  if (threadIdx.x < E_) base[threadIdx.x] = 0;
  __syncthreads();
  for (int c = 0; c < total; c += 256) {
    int e = eid[c + threadIdx.x];
    se[threadIdx.x] = e;
    __syncthreads();
    int rank = 0;
    for (int j = 0; j < (int)threadIdx.x; ++j) rank += (se[j] == e) ? 1 : 0;
    pos[c + threadIdx.x] = base[e] + rank;
    __syncthreads();
    if (threadIdx.x < E_) {
      int cnt = 0;
      for (int j = 0; j < 256; ++j) cnt += (se[j] == (int)threadIdx.x) ? 1 : 0;
      base[threadIdx.x] += cnt;
    }
    __syncthreads();
  }
}

// ---------------------------------------------------------------------------
// Dispatch: copy bf16 token row into its expert slot (drop if over capacity).
// One block per (token,k); 256 threads x 8B = 2048B = 1024 bf16.
// ---------------------------------------------------------------------------
__global__ void __launch_bounds__(256)
dispatch_tokens(const u16* __restrict__ xb, const int* __restrict__ eid,
                const int* __restrict__ pos, u16* __restrict__ disp)
{
  int i = blockIdx.x;
  int p = pos[i];
  if (p >= CAP_) return;
  int e   = eid[i];
  int tok = i >> 1;   // / KTOP_
  const uint2* src = (const uint2*)(xb + (size_t)tok * C_);
  uint2*       dst = (uint2*)(disp + ((size_t)e * CAP_ + (size_t)p) * C_);
  dst[threadIdx.x] = src[threadIdx.x];
}

// ---------------------------------------------------------------------------
// Combine: out[token] = sum_k keep_k * prob_k * eo[e_k, gpos_k]  (pure gather)
// ---------------------------------------------------------------------------
__global__ void __launch_bounds__(256)
combine_tokens(const float* __restrict__ eo, const int* __restrict__ eid,
               const int* __restrict__ pos, const float* __restrict__ tp,
               float* __restrict__ out)
{
  int n  = blockIdx.x;
  int i0 = 2 * n, i1 = 2 * n + 1;
  int e0 = eid[i0], e1 = eid[i1];
  int p0 = pos[i0], p1 = pos[i1];
  float w0 = tp[i0] * ((p0 < CAP_) ? 1.f : 0.f);
  float w1 = tp[i1] * ((p1 < CAP_) ? 1.f : 0.f);
  int g0 = (p0 < CAP_ - 1) ? p0 : (CAP_ - 1);
  int g1 = (p1 < CAP_ - 1) ? p1 : (CAP_ - 1);
  const float* r0 = eo + ((size_t)e0 * CAP_ + (size_t)g0) * C_;
  const float* r1 = eo + ((size_t)e1 * CAP_ + (size_t)g1) * C_;
  float* o = out + (size_t)n * C_;
  for (int c = threadIdx.x; c < C_; c += 256)
    o[c] = w0 * r0[c] + w1 * r1[c];
}

// ---------------------------------------------------------------------------
extern "C" void kernel_launch(void* const* d_in, const int* in_sizes, int n_in,
                              void* d_out, int out_size, void* d_ws, size_t ws_size,
                              hipStream_t stream)
{
  (void)in_sizes; (void)n_in; (void)out_size; (void)ws_size;
  const float* x   = (const float*)d_in[0];
  const float* rw1 = (const float*)d_in[1];
  const float* rb1 = (const float*)d_in[2];
  const float* rw2 = (const float*)d_in[3];
  const float* rb2 = (const float*)d_in[4];
  const float* rw3 = (const float*)d_in[5];
  const float* rb3 = (const float*)d_in[6];
  const float* ew1 = (const float*)d_in[7];
  const float* eb1 = (const float*)d_in[8];
  const float* ew2 = (const float*)d_in[9];
  const float* eb2 = (const float*)d_in[10];
  float* out = (float*)d_out;

  // Workspace layout (~430 MB) with deliberate aliasing:
  //   bigA: h1 (router hidden 1, 64MB) later reused as eh (128MB)
  //   bigB: h2 (64MB bf16) later reused as eo (64MB f32) — sizes match exactly
  char*  base = (char*)d_ws;
  size_t off  = 0;
  auto alloc = [&](size_t bytes) -> void* {
    void* p = base + off;
    off = (off + bytes + 255) & ~(size_t)255;
    return p;
  };
  u16*   xb    = (u16*)alloc((size_t)NTOK_ * C_ * 2);
  u16*   rw1t  = (u16*)alloc((size_t)C_ * H_ * 2);            // [H][C] bf16
  u16*   rw2t  = (u16*)alloc((size_t)H_ * H_ * 2);            // [H][H] bf16 (transposed)
  u16*   ew1t  = (u16*)alloc((size_t)E_ * C_ * H_ * 2);       // [E][H][C] bf16
  u16*   ew2t  = (u16*)alloc((size_t)E_ * H_ * C_ * 2);       // [E][C][H] bf16
  u16*   bigA  = (u16*)alloc((size_t)E_ * CAP_ * H_ * 2);     // h1 / eh
  void*  bigB  =        alloc((size_t)NTOK_ * H_ * 2);        // h2 / eo
  u16*   disp  = (u16*)alloc((size_t)E_ * CAP_ * C_ * 2);
  int*   tkid  = (int*)alloc((size_t)NTOK_ * KTOP_ * 4);
  float* tkp   = (float*)alloc((size_t)NTOK_ * KTOP_ * 4);
  int*   pos   = (int*)alloc((size_t)NTOK_ * KTOP_ * 4);
  u16*   h1 = bigA;
  u16*   eh = bigA;
  u16*   h2 = (u16*)bigB;
  float* eo = (float*)bigB;

  // 1) activation f32->bf16; weights f32->bf16 WITH transpose to [N][K]
  {
    int n4 = (int)((size_t)NTOK_ * C_ / 4);
    cvt_f32_bf16<<<(n4 + 255) / 256, 256, 0, stream>>>(x, xb, n4);
  }
  cvt_transpose_f32_bf16<<<dim3(H_ / 32, C_ / 32, 1),  256, 0, stream>>>(rw1, rw1t, C_, H_);
  cvt_transpose_f32_bf16<<<dim3(H_ / 32, H_ / 32, 1),  256, 0, stream>>>(rw2, rw2t, H_, H_);
  cvt_transpose_f32_bf16<<<dim3(H_ / 32, C_ / 32, E_), 256, 0, stream>>>(ew1, ew1t, C_, H_);
  cvt_transpose_f32_bf16<<<dim3(C_ / 32, H_ / 32, E_), 256, 0, stream>>>(ew2, ew2t, H_, C_);

  dim3 blk(256);
  // 2) router h1 = relu(x @ rw1 + rb1)   [8192 x 4096 x 1024]
  gemm_bf16_wmma<true, true><<<dim3(H_ / 128, NTOK_ / 128, 1), blk, 0, stream>>>(
      xb, rw1t, rb1, h1, NTOK_, H_, C_, 0, 0, 0, 0);
  // 3) router h2 = relu(h1 @ rw2 + rb2)  [8192 x 4096 x 4096]
  gemm_bf16_wmma<true, true><<<dim3(H_ / 128, NTOK_ / 128, 1), blk, 0, stream>>>(
      h1, rw2t, rb2, h2, NTOK_, H_, H_, 0, 0, 0, 0);

  // 4) gating: logits -> softmax -> top-2
  gate_topk<<<NTOK_, 256, 0, stream>>>(h2, rw3, rb3, tkid, tkp);

  // 5) deterministic per-expert positions (flat order, matches reference)
  compute_pos<<<1, 256, 0, stream>>>(tkid, pos, NTOK_ * KTOP_);

  // 6) zero + dispatch expert buffers
  {
    size_t dispU = (size_t)E_ * CAP_ * C_ / 2;   // uint count
    zero_u32<<<(unsigned)((dispU + 255) / 256), 256, 0, stream>>>((unsigned int*)disp, (int)dispU);
  }
  dispatch_tokens<<<NTOK_ * KTOP_, 256, 0, stream>>>(xb, tkid, pos, disp);

  // 7) expert GEMM1: eh = relu(disp @ ew1 + eb1), batched over 8 experts
  gemm_bf16_wmma<true, true><<<dim3(H_ / 128, CAP_ / 128, E_), blk, 0, stream>>>(
      disp, ew1t, eb1, eh, CAP_, H_, C_,
      (long long)CAP_ * C_, (long long)C_ * H_, (long long)H_, (long long)CAP_ * H_);
  // 8) expert GEMM2: eo = eh @ ew2 + eb2 (f32 out), batched
  gemm_bf16_wmma<false, false><<<dim3(C_ / 128, CAP_ / 128, E_), blk, 0, stream>>>(
      eh, ew2t, eb2, eo, CAP_, C_, H_,
      (long long)CAP_ * H_, (long long)H_ * C_, (long long)C_, (long long)CAP_ * C_);

  // 9) combine -> d_out (pure per-token gather, deterministic)
  combine_tokens<<<NTOK_, 256, 0, stream>>>(eo, tkid, pos, tkp, out);
}
